// PointTransformer_MAE_36919538876840
// MI455X (gfx1250) — compile-verified, matrix operand order checked by reading, not run on hardware
//
#include <hip/hip_runtime.h>

// PointTransformer-MAE encoder for MI455X (gfx1250, wave32, WMMA).
// FPS -> KNN+group -> conv1(3->128)+BNstats -> [BN,ReLU,GEMM2,maxcat,GEMM3]+BNstats
//     -> [BN,ReLU,GEMM4,maxpool] -> reduce GEMM.
// GEMMs: v_wmma_f32_16x16x32_f16.  Activation staging: global_load_async_to_lds_b128.

#define BB 8
#define NN 16384
#define GG 256
#define KK 32
#define MGRP (BB*GG)      /* 2048 groups */
#define ROWS (MGRP*KK)    /* 65536 rows through the encoder */
#define C1 128
#define C2 256
#define C3 512
#define CE 384
#define CT 384

typedef __attribute__((ext_vector_type(16))) _Float16 v16h;
typedef __attribute__((ext_vector_type(8)))  float    v8f;
typedef __attribute__((ext_vector_type(2)))  _Float16 h2;
typedef __attribute__((ext_vector_type(4)))  int      v4i;

// ---- CDNA5 async global->LDS copy (ASYNCcnt-tracked DMA, 16B per lane) -----
__device__ __forceinline__ void async_copy_b128(const void* g, void* l) {
#if __has_builtin(__builtin_amdgcn_global_load_async_to_lds_b128)
    __builtin_amdgcn_global_load_async_to_lds_b128((v4i*)g, (v4i*)l, 0, 0);
#else
    const unsigned long long ga = (unsigned long long)(size_t)g;
    const unsigned int la = (unsigned int)(size_t)l;   // flat LDS addr[31:0] == LDS offset
    asm volatile("global_load_async_to_lds_b128 %0, %1, off" :: "v"(la), "v"(ga) : "memory");
#endif
}

__device__ __forceinline__ void wait_async0() {
#if __has_builtin(__builtin_amdgcn_s_wait_asynccnt)
    __builtin_amdgcn_s_wait_asynccnt(0);
#else
    asm volatile("s_wait_asynccnt 0x0" ::: "memory");
#endif
}

// ---- WMMA fragment helpers -------------------------------------------------
// 16-bit A fragment layout (ISA 7.12.2): m = lane&15, half = lane>>4;
// f16 pair v (0..7): k = 2v + (v>=4 ? 8 : 0) + 8*half, +0/+1.  (Compiler merges
// these into ds_load_b128 / global_load_b128 — verified in round-1 disasm.)
// B fragment mirrored (lane holds column n, same K packing); we compute f @ W^T,
// so B[k][n] = W[n][k] with W row-major -> contiguous 16B reads.
__device__ __forceinline__ v16h ld_frag(const _Float16* __restrict__ p, int ld,
                                        int row0, int k0, int lane) {
    const int m = row0 + (lane & 15);
    const int hh = lane >> 4;
    const h2* rp = (const h2*)(p + (size_t)m * ld + k0);
    v16h r;
    #pragma unroll
    for (int v = 0; v < 8; ++v) {
        const int kb = 2 * v + ((v >= 4) ? 8 : 0) + 8 * hh;  // even
        h2 t = rp[kb >> 1];
        r[2 * v] = t[0];
        r[2 * v + 1] = t[1];
    }
    return r;
}

__device__ __forceinline__ v8f wmma32(v16h a, v16h b, v8f c) {
    return __builtin_amdgcn_wmma_f32_16x16x32_f16(false, a, false, b,
                                                  (short)0, c, false, false);
}

// ---- utility kernels -------------------------------------------------------
__global__ void k_zerof(float* __restrict__ p, int n) {
    int i = blockIdx.x * blockDim.x + threadIdx.x;
    if (i < n) p[i] = 0.0f;
}

__global__ void k_tohalf(const float* __restrict__ s, _Float16* __restrict__ d, int n) {
    int i = blockIdx.x * blockDim.x + threadIdx.x;
    if (i < n) d[i] = (_Float16)s[i];
}

__global__ void k_stats(const float* __restrict__ sum, const float* __restrict__ sq,
                        float* __restrict__ mean, float* __restrict__ rstd,
                        int C, float inv_count) {
    int c = blockIdx.x * blockDim.x + threadIdx.x;
    if (c < C) {
        float m = sum[c] * inv_count;
        float v = sq[c] * inv_count - m * m;
        mean[c] = m;
        rstd[c] = rsqrtf(v + 1e-5f);
    }
}

// ---- Farthest point sampling: 1 block per batch, 16 points/thread ----------
__global__ __launch_bounds__(1024) void k_fps(const float* __restrict__ pts,
                                              int* __restrict__ cidx) {
    const int b = blockIdx.x, t = threadIdx.x;
    const float* P = pts + (size_t)b * NN * 3;
    float px[16], py[16], pz[16], dist[16];
    #pragma unroll
    for (int j = 0; j < 16; ++j) {
        int n = j * 1024 + t;
        px[j] = P[n * 3 + 0]; py[j] = P[n * 3 + 1]; pz[j] = P[n * 3 + 2];
        dist[j] = 1e10f;
    }
    __shared__ float rv[1024];
    __shared__ int   ri[1024];
    __shared__ int   s_far;
    if (t == 0) s_far = 0;
    __syncthreads();
    for (int i = 0; i < GG; ++i) {
        const int far = s_far;
        if (t == 0) cidx[b * GG + i] = far;
        const float cx = P[far * 3 + 0], cy = P[far * 3 + 1], cz = P[far * 3 + 2];
        float bv = -1.0f; int bi = 0;
        #pragma unroll
        for (int j = 0; j < 16; ++j) {
            float dx = px[j] - cx, dy = py[j] - cy, dz = pz[j] - cz;
            float d = dx * dx + dy * dy + dz * dz;
            float nd = fminf(dist[j], d);
            dist[j] = nd;
            if (nd > bv) { bv = nd; bi = j * 1024 + t; }
        }
        rv[t] = bv; ri[t] = bi;
        __syncthreads();
        for (int s = 512; s > 0; s >>= 1) {
            if (t < s) {
                float ov = rv[t + s]; int oi = ri[t + s];
                if (ov > rv[t] || (ov == rv[t] && oi < ri[t])) { rv[t] = ov; ri[t] = oi; }
            }
            __syncthreads();
        }
        if (t == 0) s_far = ri[0];
        __syncthreads();
    }
}

// ---- KNN + grouping: 1 block per center, full d2 array in WGP LDS ----------
__global__ __launch_bounds__(256) void k_knn(const float* __restrict__ pts,
                                             const int* __restrict__ cidx,
                                             float* __restrict__ pg) {
    const int gid = blockIdx.x;
    const int b = gid >> 8, t = threadIdx.x;
    const float* P = pts + (size_t)b * NN * 3;
    __shared__ float d2[NN];        // 64KB (CDNA5 WGP LDS = 320KB)
    __shared__ float rv[256];
    __shared__ int   ri[256];
    __shared__ int   nlist[KK];
    const int cid = cidx[gid];
    const float cx = P[cid * 3 + 0], cy = P[cid * 3 + 1], cz = P[cid * 3 + 2];
    for (int j = 0; j < 64; ++j) {
        int n = j * 256 + t;
        float dx = P[n * 3 + 0] - cx, dy = P[n * 3 + 1] - cy, dz = P[n * 3 + 2] - cz;
        d2[n] = dx * dx + dy * dy + dz * dz;
    }
    __syncthreads();
    for (int kk = 0; kk < KK; ++kk) {
        float bv = 3.0e38f; int bi = 0;
        for (int j = 0; j < 64; ++j) {
            int n = j * 256 + t;
            float v = d2[n];
            if (v < bv) { bv = v; bi = n; }
        }
        rv[t] = bv; ri[t] = bi;
        __syncthreads();
        for (int s = 128; s > 0; s >>= 1) {
            if (t < s) {
                float ov = rv[t + s]; int oi = ri[t + s];
                if (ov < rv[t] || (ov == rv[t] && oi < ri[t])) { rv[t] = ov; ri[t] = oi; }
            }
            __syncthreads();
        }
        if (t == 0) { int w = ri[0]; nlist[kk] = w; d2[w] = 3.0e38f; }
        __syncthreads();
    }
    if (t < KK * 3) {
        int k = t / 3, d = t % 3;
        int n = nlist[k];
        float cv = (d == 0) ? cx : ((d == 1) ? cy : cz);
        pg[(size_t)gid * (KK * 3) + t] = P[n * 3 + d] - cv;
    }
}

// ---- conv1 (3->128) + BN1 partial stats ------------------------------------
__global__ __launch_bounds__(128) void k_enc1(const float* __restrict__ pg,
                                              const float* __restrict__ w1,
                                              const float* __restrict__ b1,
                                              float* __restrict__ a1,
                                              float* __restrict__ gsum,
                                              float* __restrict__ gsq) {
    const int g = blockIdx.x, c = threadIdx.x;
    __shared__ float sp[KK * 3];
    if (c < KK * 3) sp[c] = pg[(size_t)g * (KK * 3) + c];
    __syncthreads();
    const float wa = w1[c * 3 + 0], wb = w1[c * 3 + 1], wc = w1[c * 3 + 2], bc = b1[c];
    float s = 0.0f, q = 0.0f;
    #pragma unroll 4
    for (int m = 0; m < KK; ++m) {
        float x = sp[m * 3 + 0] * wa + sp[m * 3 + 1] * wb + sp[m * 3 + 2] * wc + bc;
        a1[((size_t)g * KK + m) * C1 + c] = x;
        s += x; q += x * x;
    }
    atomicAdd(&gsum[c], s);
    atomicAdd(&gsq[c], q);
}

// ---- BN1+ReLU -> GEMM2 (WMMA) -> max-cat -> GEMM3 (WMMA) + BN3 stats -------
__global__ __launch_bounds__(128) void k_enc2(const float* __restrict__ a1,
    const float* __restrict__ mean1, const float* __restrict__ rstd1,
    const float* __restrict__ g1, const float* __restrict__ be1,
    const _Float16* __restrict__ w2h, const float* __restrict__ b2,
    const _Float16* __restrict__ w3h, const float* __restrict__ b3,
    _Float16* __restrict__ a3h, float* __restrict__ gsum3, float* __restrict__ gsq3) {
    const int g = blockIdx.x, tid = threadIdx.x;
    const int wave = tid >> 5, lane = tid & 31;
    __shared__ __align__(16) _Float16 h1[KK * C1];   // 8KB
    __shared__ __align__(16) _Float16 cat[KK * C3];  // 32KB: [:,0:256)=fg, [:,256:512)=f
    __shared__ float lsum[C3], lsq[C3];

    // Phase 0: async-DMA the 16KB a1 block into the (currently dead) cat region.
    {
        const char* gsrc = (const char*)(a1 + (size_t)g * KK * C1);
        char* ldst = (char*)cat;
        #pragma unroll
        for (int i = 0; i < (KK * C1 * 4) / (16 * 128); ++i) {   // 8 chunks/thread
            int o = (i * 128 + tid) * 16;
            async_copy_b128(gsrc + o, ldst + o);
        }
    }
    for (int i = tid; i < C3; i += 128) { lsum[i] = 0.0f; lsq[i] = 0.0f; }
    wait_async0();
    __syncthreads();
    // Phase 1: BN1 + ReLU + f16 pack, cat(f32 view) -> h1.
    {
        const float* stagef = (const float*)cat;
        for (int i = tid; i < KK * C1; i += 128) {
            int c = i & (C1 - 1);
            float x = stagef[i];
            x = (x - mean1[c]) * rstd1[c] * g1[c] + be1[c];
            h1[i] = (_Float16)fmaxf(x, 0.0f);
        }
    }
    __syncthreads();
    // GEMM2: (32 x 128) @ w2^T -> 32 x 256, bias, write f16 into cat[:,256:]
    for (int tI = wave; tI < 2 * (C2 / 16); tI += 4) {
        const int mt = tI >> 4, nt = tI & 15;
        v8f acc = {};
        #pragma unroll
        for (int kk = 0; kk < C1 / 32; ++kk) {
            v16h a = ld_frag(h1, C1, mt * 16, kk * 32, lane);
            v16h bf = ld_frag(w2h, C1, nt * 16, kk * 32, lane);
            acc = wmma32(a, bf, acc);
        }
        const int col = nt * 16 + (lane & 15);
        const float bias = b2[col];
        const int rbase = mt * 16 + 8 * (lane >> 4);
        #pragma unroll
        for (int r = 0; r < 8; ++r)
            cat[(rbase + r) * C3 + C2 + col] = (_Float16)(acc[r] + bias);
    }
    __syncthreads();
    // per-group channel max over K rows, broadcast into cat[:,0:256)
    for (int c = tid; c < C2; c += 128) {
        float mx = -3.0e38f;
        for (int m = 0; m < KK; ++m) mx = fmaxf(mx, (float)cat[m * C3 + C2 + c]);
        _Float16 mh = (_Float16)mx;
        for (int m = 0; m < KK; ++m) cat[m * C3 + c] = mh;
    }
    __syncthreads();
    // GEMM3: (32 x 512) @ w3^T -> 32 x 512 (pre-BN, +bias), store f16 + stats
    for (int tI = wave; tI < 2 * (C3 / 16); tI += 4) {
        const int mt = tI >> 5, nt = tI & 31;
        v8f acc = {};
        #pragma unroll 4
        for (int kk = 0; kk < C3 / 32; ++kk) {
            if (kk + 1 < C3 / 32)
                __builtin_prefetch(w3h + (size_t)(nt * 16 + (lane & 15)) * C3 + (kk + 1) * 32, 0, 3);
            v16h a = ld_frag(cat, C3, mt * 16, kk * 32, lane);
            v16h bf = ld_frag(w3h, C3, nt * 16, kk * 32, lane);
            acc = wmma32(a, bf, acc);
        }
        const int col = nt * 16 + (lane & 15);
        const float bias = b3[col];
        const int rbase = mt * 16 + 8 * (lane >> 4);
        float s = 0.0f, q = 0.0f;
        #pragma unroll
        for (int r = 0; r < 8; ++r) {
            float x = acc[r] + bias;
            a3h[((size_t)g * KK + rbase + r) * C3 + col] = (_Float16)x;
            s += x; q += x * x;
        }
        atomicAdd(&lsum[col], s);
        atomicAdd(&lsq[col], q);
    }
    __syncthreads();
    for (int c = tid; c < C3; c += 128) {
        atomicAdd(&gsum3[c], lsum[c]);
        atomicAdd(&gsq3[c], lsq[c]);
    }
}

// ---- BN3+ReLU -> GEMM4 (WMMA) -> max-pool -> token (f16) -------------------
__global__ __launch_bounds__(128) void k_enc3(const _Float16* __restrict__ a3h,
    const float* __restrict__ mean3, const float* __restrict__ rstd3,
    const float* __restrict__ g3, const float* __restrict__ be3,
    const _Float16* __restrict__ w4h, const float* __restrict__ b4,
    _Float16* __restrict__ tokh) {
    const int g = blockIdx.x, tid = threadIdx.x;
    const int wave = tid >> 5, lane = tid & 31;
    __shared__ __align__(16) _Float16 h3[KK * C3];    // 32KB
    __shared__ __align__(16) _Float16 u[KK * C3];     // 32KB: stage (ph.0-1) / o4 (ph.2+)
    _Float16* o4 = u;                                 // 32 x 384 (24KB) after GEMM4

    // Phase 0: async-DMA the 32KB raw a3h block into the stage region.
    {
        const char* gsrc = (const char*)(a3h + (size_t)g * KK * C3);
        char* ldst = (char*)u;
        #pragma unroll
        for (int i = 0; i < (KK * C3 * 2) / (16 * 128); ++i) {   // 16 chunks/thread
            int o = (i * 128 + tid) * 16;
            async_copy_b128(gsrc + o, ldst + o);
        }
    }
    wait_async0();
    __syncthreads();
    // Phase 1: BN3 + ReLU, stage -> h3.
    for (int i = tid; i < KK * C3; i += 128) {
        int c = i & (C3 - 1);
        float x = (float)u[i];
        x = (x - mean3[c]) * rstd3[c] * g3[c] + be3[c];
        h3[i] = (_Float16)fmaxf(x, 0.0f);
    }
    __syncthreads();
    for (int tI = wave; tI < 2 * (CE / 16); tI += 4) {
        const int mt = tI / (CE / 16), nt = tI % (CE / 16);
        v8f acc = {};
        #pragma unroll 4
        for (int kk = 0; kk < C3 / 32; ++kk) {
            v16h a = ld_frag(h3, C3, mt * 16, kk * 32, lane);
            v16h bf = ld_frag(w4h, C3, nt * 16, kk * 32, lane);
            acc = wmma32(a, bf, acc);
        }
        const int col = nt * 16 + (lane & 15);
        const float bias = b4[col];
        const int rbase = mt * 16 + 8 * (lane >> 4);
        #pragma unroll
        for (int r = 0; r < 8; ++r)
            o4[(rbase + r) * CE + col] = (_Float16)(acc[r] + bias);
    }
    __syncthreads();
    for (int c = tid; c < CE; c += 128) {
        float mx = -3.0e38f;
        for (int m = 0; m < KK; ++m) mx = fmaxf(mx, (float)o4[m * CE + c]);
        tokh[(size_t)g * CE + c] = (_Float16)mx;
    }
}

// ---- reduce_dim: tok (2048x384) @ wr^T + br -> out, one wave per 16x16 tile -
__global__ __launch_bounds__(32) void k_reduce(const _Float16* __restrict__ tokh,
                                               const _Float16* __restrict__ wrh,
                                               const float* __restrict__ br,
                                               float* __restrict__ out) {
    const int lane = threadIdx.x;
    const int row0 = blockIdx.x * 16;
    const int n0 = blockIdx.y * 16;
    v8f acc = {};
    #pragma unroll
    for (int kk = 0; kk < CE / 32; ++kk) {
        v16h a = ld_frag(tokh, CE, row0, kk * 32, lane);
        v16h bf = ld_frag(wrh, CE, n0, kk * 32, lane);
        acc = wmma32(a, bf, acc);
    }
    const int col = n0 + (lane & 15);
    const float bias = br[col];
    const int rbase = row0 + 8 * (lane >> 4);
    #pragma unroll
    for (int r = 0; r < 8; ++r)
        out[(size_t)(rbase + r) * CT + col] = acc[r] + bias;
}

// ---- host-side orchestration ----------------------------------------------
extern "C" void kernel_launch(void* const* d_in, const int* in_sizes, int n_in,
                              void* d_out, int out_size, void* d_ws, size_t ws_size,
                              hipStream_t stream) {
    (void)in_sizes; (void)n_in; (void)out_size; (void)ws_size;
    const float* points = (const float*)d_in[0];
    const float* w1 = (const float*)d_in[1];
    const float* b1 = (const float*)d_in[2];
    const float* g1 = (const float*)d_in[3];
    const float* be1 = (const float*)d_in[4];
    const float* w2 = (const float*)d_in[5];
    const float* b2 = (const float*)d_in[6];
    const float* w3 = (const float*)d_in[7];
    const float* b3 = (const float*)d_in[8];
    const float* g3 = (const float*)d_in[9];
    const float* be3 = (const float*)d_in[10];
    const float* w4 = (const float*)d_in[11];
    const float* b4 = (const float*)d_in[12];
    const float* wr = (const float*)d_in[13];
    const float* br = (const float*)d_in[14];
    float* out = (float*)d_out;

    char* ws = (char*)d_ws;
    size_t off = 0;
    auto take = [&](size_t bytes) -> char* {
        char* p = ws + off;
        off += (bytes + 255) & ~(size_t)255;
        return p;
    };
    int*      cidx = (int*)take((size_t)MGRP * sizeof(int));
    float*    pg   = (float*)take((size_t)MGRP * KK * 3 * sizeof(float));
    float*    a1   = (float*)take((size_t)ROWS * C1 * sizeof(float));      // 33.5 MB
    _Float16* a3h  = (_Float16*)take((size_t)ROWS * C3 * 2);               // 67 MB (f16)
    _Float16* tokh = (_Float16*)take((size_t)MGRP * CE * 2);
    _Float16* w2h  = (_Float16*)take((size_t)C2 * C1 * 2);
    _Float16* w3h  = (_Float16*)take((size_t)C3 * C3 * 2);
    _Float16* w4h  = (_Float16*)take((size_t)CE * C3 * 2);
    _Float16* wrh  = (_Float16*)take((size_t)CT * CE * 2);
    float*    stats = (float*)take(2560 * sizeof(float));
    float* gsum1 = stats;        float* gsq1 = stats + 128;
    float* mean1 = stats + 256;  float* rstd1 = stats + 384;
    float* gsum3 = stats + 512;  float* gsq3 = stats + 1024;
    float* mean3 = stats + 1536; float* rstd3 = stats + 2048;

    k_zerof<<<(2560 + 255) / 256, 256, 0, stream>>>(stats, 2560);
    k_tohalf<<<(C2 * C1 + 255) / 256, 256, 0, stream>>>(w2, w2h, C2 * C1);
    k_tohalf<<<(C3 * C3 + 255) / 256, 256, 0, stream>>>(w3, w3h, C3 * C3);
    k_tohalf<<<(CE * C3 + 255) / 256, 256, 0, stream>>>(w4, w4h, CE * C3);
    k_tohalf<<<(CT * CE + 255) / 256, 256, 0, stream>>>(wr, wrh, CT * CE);

    k_fps<<<BB, 1024, 0, stream>>>(points, cidx);
    k_knn<<<MGRP, 256, 0, stream>>>(points, cidx, pg);
    k_enc1<<<MGRP, 128, 0, stream>>>(pg, w1, b1, a1, gsum1, gsq1);
    k_stats<<<1, 128, 0, stream>>>(gsum1, gsq1, mean1, rstd1, C1, 1.0f / (float)ROWS);
    k_enc2<<<MGRP, 128, 0, stream>>>(a1, mean1, rstd1, g1, be1, w2h, b2, w3h, b3,
                                     a3h, gsum3, gsq3);
    k_stats<<<4, 128, 0, stream>>>(gsum3, gsq3, mean3, rstd3, C3, 1.0f / (float)ROWS);
    k_enc3<<<MGRP, 128, 0, stream>>>(a3h, mean3, rstd3, g3, be3, w4h, b4, tokh);
    k_reduce<<<dim3(MGRP / 16, CT / 16), 32, 0, stream>>>(tokh, wrh, br, out);
}
